// MiniGPT4AttentionCam_74577812128653
// MI455X (gfx1250) — compile-verified, matrix operand order checked by reading, not run on hardware
//
#include <hip/hip_runtime.h>
#include <math.h>

// ---------------- problem constants ----------------
#define T_SEQ   2048
#define DMODEL  2048
#define NHEAD   16
#define HDIM    128
#define SB      204      // int(0.1*2048)
#define RB      1024     // int(0.5*2048)
#define NVALID  1228     // SB + RB
#define SCALE_QK 0.088388347648318447f   // 1/sqrt(128)

typedef float  v8f  __attribute__((ext_vector_type(8)));
typedef __bf16 v4bf __attribute__((ext_vector_type(4)));
typedef __bf16 v8bf __attribute__((ext_vector_type(8)));
typedef __bf16 v16bf __attribute__((ext_vector_type(16)));

static __device__ __forceinline__ v8f wmma_bf16(v16bf a, v16bf b, v8f c) {
  // D(16x16,f32) = A(16x32,bf16) x B(32x16,bf16) + C
  return __builtin_amdgcn_wmma_f32_16x16x32_bf16(false, a, false, b, (short)0, c, false, false);
}

// 16-bit A/B fragment: 8 contiguous K at p, 8 more at p+16
static __device__ __forceinline__ v16bf ld_frag(const __bf16* p) {
  v8bf a = *(const v8bf*)(p);
  v8bf b = *(const v8bf*)(p + 16);
  return __builtin_shufflevector(a, b, 0,1,2,3,4,5,6,7,8,9,10,11,12,13,14,15);
}

static __device__ __forceinline__ void split4(float4 v, v4bf& h, v4bf& l) {
  h[0] = (__bf16)v.x; l[0] = (__bf16)(v.x - (float)h[0]);
  h[1] = (__bf16)v.y; l[1] = (__bf16)(v.y - (float)h[1]);
  h[2] = (__bf16)v.z; l[2] = (__bf16)(v.z - (float)h[2]);
  h[3] = (__bf16)v.w; l[3] = (__bf16)(v.w - (float)h[3]);
}

static __device__ __forceinline__ float fin_clamp(float v) {
  if (v != v)        return 0.0f;
  if (v >  1.0e38f)  return  1.0e4f;
  if (v < -1.0e38f)  return -1.0e4f;
  return v;
}

static __device__ __forceinline__ float rmax16(float v) {
  #pragma unroll
  for (int m = 1; m < 16; m <<= 1) v = fmaxf(v, __shfl_xor(v, m, 32));
  return v;
}
static __device__ __forceinline__ float rsum16(float v) {
  #pragma unroll
  for (int m = 1; m < 16; m <<= 1) v += __shfl_xor(v, m, 32);
  return v;
}

// ---------------- async global->LDS copy (CDNA5), with sync fallback ----------------
#if __has_builtin(__builtin_amdgcn_global_load_async_to_lds_b128)
#define HAS_ASYNC_LDS 1
typedef int gv4i __attribute__((vector_size(16)));
typedef __attribute__((address_space(1))) gv4i as1_v4i;   // global
typedef __attribute__((address_space(3))) gv4i as3_v4i;   // LDS
static __device__ __forceinline__ void async_cp16(void* l, const void* g) {
  __builtin_amdgcn_global_load_async_to_lds_b128((as1_v4i*)g, (as3_v4i*)l, 0, 0);
}
static __device__ __forceinline__ void wait_async0() {
#if __has_builtin(__builtin_amdgcn_s_wait_asynccnt)
  __builtin_amdgcn_s_wait_asynccnt(0);
#else
  asm volatile("s_wait_asynccnt 0" ::: "memory");
#endif
}
#else
#define HAS_ASYNC_LDS 0
static __device__ __forceinline__ void async_cp16(void* l, const void* g) {
  *(float4*)l = *(const float4*)g;     // 16B sync copy fallback
}
static __device__ __forceinline__ void wait_async0() {}
#endif

// ---------------- kernel 1: nan_to_num(optional) + bf16 hi/lo split ----------------
__global__ void split_kernel(const float* __restrict__ x, __bf16* __restrict__ hi,
                             __bf16* __restrict__ lo, int n, int do_clamp) {
  int i = blockIdx.x * blockDim.x + threadIdx.x;
  if (i < n) {
    float v = x[i];
    if (do_clamp) v = fin_clamp(v);
    __bf16 h = (__bf16)v;
    hi[i] = h;
    lo[i] = (__bf16)(v - (float)h);
  }
}

// ---------------- GEMM: C = A * W^T, bf16x3, async double-buffered LDS ----------------
// Operands pre-split to global bf16 hi/lo.  mode 0: scatter [H][T][HDIM]; 1: [T][D]+clamp.
__global__ __launch_bounds__(256)
void gemm_bf16x3_async_kernel(const __bf16* __restrict__ Ahi_g, const __bf16* __restrict__ Alo_g,
                              const __bf16* __restrict__ Whi_g, const __bf16* __restrict__ Wlo_g,
                              float* __restrict__ out, int mode) {
  __shared__ __bf16 Abuf[2][2][64 * 40];    // [buf][hi/lo], 64 rows x BK=32, stride 40
  __shared__ __bf16 Wbuf[2][2][128 * 40];

  const int tid  = threadIdx.x;
  const int wid  = tid >> 5;
  const int lane = tid & 31;
  const int ln   = lane & 15;
  const int hf   = lane >> 4;
  const int wr   = wid >> 2;       // 0..1
  const int wc   = wid & 3;        // 0..3
  const int m0   = blockIdx.y * 64;
  const int n0   = blockIdx.x * 128;

  // per-thread 16B-chunk assignments: A = 256 chunks (1/thread/array), W = 512 (2/thread/array)
  const int arow = tid >> 2, ac = (tid & 3) * 8;       // 8 bf16 = 16B

  v8f acc[2][2];
  #pragma unroll
  for (int i = 0; i < 2; ++i)
    #pragma unroll
    for (int j = 0; j < 2; ++j)
      #pragma unroll
      for (int r = 0; r < 8; ++r) acc[i][j][r] = 0.0f;

  auto issue = [&](int buf, int kt) {
    async_cp16(&Abuf[buf][0][arow * 40 + ac], Ahi_g + (size_t)(m0 + arow) * DMODEL + kt + ac);
    async_cp16(&Abuf[buf][1][arow * 40 + ac], Alo_g + (size_t)(m0 + arow) * DMODEL + kt + ac);
    #pragma unroll
    for (int i = 0; i < 2; ++i) {
      int idx = tid + i * 256, rw = idx >> 2, cw = (idx & 3) * 8;
      async_cp16(&Wbuf[buf][0][rw * 40 + cw], Whi_g + (size_t)(n0 + rw) * DMODEL + kt + cw);
      async_cp16(&Wbuf[buf][1][rw * 40 + cw], Wlo_g + (size_t)(n0 + rw) * DMODEL + kt + cw);
    }
  };

  issue(0, 0);
  for (int i = 0; i < DMODEL / 32; ++i) {
    const int kt  = i * 32;
    const int buf = i & 1;
    wait_async0();          // own async copies for buf complete
    __syncthreads();        // all waves' copies complete; all done computing i-1

    if (kt + 32 < DMODEL) issue(buf ^ 1, kt + 32);   // overlap next tile with compute

    const __bf16* Ah = Abuf[buf][0];
    const __bf16* Al = Abuf[buf][1];
    const __bf16* Wh = Wbuf[buf][0];
    const __bf16* Wl = Wbuf[buf][1];
    const int ab = (wr * 32 + ln) * 40 + 8 * hf;
    const int bb = (wc * 32 + ln) * 40 + 8 * hf;
    v16bf a0h = ld_frag(&Ah[ab]),           a0l = ld_frag(&Al[ab]);
    v16bf a1h = ld_frag(&Ah[ab + 16 * 40]), a1l = ld_frag(&Al[ab + 16 * 40]);
    v16bf b0h = ld_frag(&Wh[bb]),           b0l = ld_frag(&Wl[bb]);
    v16bf b1h = ld_frag(&Wh[bb + 16 * 40]), b1l = ld_frag(&Wl[bb + 16 * 40]);

    // product-major, round-robin over the 4 independent accumulators:
    // each accumulator's C-chain has distance 4 -> no copies / hazard stalls
    acc[0][0] = wmma_bf16(a0h, b0l, acc[0][0]);
    acc[0][1] = wmma_bf16(a0h, b1l, acc[0][1]);
    acc[1][0] = wmma_bf16(a1h, b0l, acc[1][0]);
    acc[1][1] = wmma_bf16(a1h, b1l, acc[1][1]);
    acc[0][0] = wmma_bf16(a0l, b0h, acc[0][0]);
    acc[0][1] = wmma_bf16(a0l, b1h, acc[0][1]);
    acc[1][0] = wmma_bf16(a1l, b0h, acc[1][0]);
    acc[1][1] = wmma_bf16(a1l, b1h, acc[1][1]);
    acc[0][0] = wmma_bf16(a0h, b0h, acc[0][0]);
    acc[0][1] = wmma_bf16(a0h, b1h, acc[0][1]);
    acc[1][0] = wmma_bf16(a1h, b0h, acc[1][0]);
    acc[1][1] = wmma_bf16(a1h, b1h, acc[1][1]);
  }

  // C layout: lane ln = column, row = r + 8*hf
  #pragma unroll
  for (int i = 0; i < 2; ++i)
    #pragma unroll
    for (int j = 0; j < 2; ++j)
      #pragma unroll
      for (int r = 0; r < 8; ++r) {
        int gm = m0 + wr * 32 + i * 16 + r + 8 * hf;
        int gn = n0 + wc * 32 + j * 16 + ln;
        float v = acc[i][j][r];
        if (mode == 1) {
          out[(size_t)gm * DMODEL + gn] = fin_clamp(v);
        } else {
          int h = gn >> 7, d = gn & 127;
          out[((size_t)h * T_SEQ + gm) * HDIM + d] = v;
        }
      }
}

// ---------------- kernel 3: RoPE on Q,K + emit bf16 hi/lo split of K ----------------
__global__ void rope_split_kernel(float* __restrict__ Q, float* __restrict__ K,
                                  __bf16* __restrict__ Khi, __bf16* __restrict__ Klo) {
  int idx = blockIdx.x * blockDim.x + threadIdx.x;       // H*T*64
  if (idx >= NHEAD * T_SEQ * 64) return;
  int j = idx & 63;
  int t = (idx >> 6) & (T_SEQ - 1);
  int h = idx >> 17;
  float invf = __expf(-(float)j * (9.210340371976184f / 64.0f)); // 10000^(-j/64)
  float ang = (float)t * invf;
  float c = cosf(ang), s = sinf(ang);
  size_t base = ((size_t)h * T_SEQ + t) * HDIM;

  float q1 = Q[base + j], q2 = Q[base + j + 64];
  Q[base + j]      = q1 * c - q2 * s;
  Q[base + j + 64] = q2 * c + q1 * s;

  float k1 = K[base + j], k2 = K[base + j + 64];
  float k1n = k1 * c - k2 * s;
  float k2n = k2 * c + k1 * s;
  K[base + j]      = k1n;
  K[base + j + 64] = k2n;
  __bf16 h1 = (__bf16)k1n; Khi[base + j]      = h1; Klo[base + j]      = (__bf16)(k1n - (float)h1);
  __bf16 h2 = (__bf16)k2n; Khi[base + j + 64] = h2; Klo[base + j + 64] = (__bf16)(k2n - (float)h2);
}

// ---------------- kernel 4: last-row softmax -> stochastic V bump (f32) ----------------
__global__ __launch_bounds__(256)
void vupdate_kernel(const float* __restrict__ Q, const float* __restrict__ K,
                    const float* __restrict__ bern, float* __restrict__ V) {
  const int h = blockIdx.x, tid = threadIdx.x;
  __shared__ float sc[NVALID];
  __shared__ float qrow[HDIM];
  __shared__ float red[256];
  __shared__ float sh_coef;
  __shared__ float ve[HDIM];

  if (tid < HDIM) qrow[tid] = Q[((size_t)h * T_SEQ + (T_SEQ - 1)) * HDIM + tid];
  __syncthreads();

  float lmax = -1e30f;
  for (int idx = tid; idx < NVALID; idx += 256) {
    int key = (idx < SB) ? idx : (T_SEQ - RB) + (idx - SB);
    const float* kr = K + ((size_t)h * T_SEQ + key) * HDIM;
    float d = 0.0f;
    #pragma unroll 8
    for (int c = 0; c < HDIM; ++c) d += qrow[c] * kr[c];
    d *= SCALE_QK;
    sc[idx] = d;
    lmax = fmaxf(lmax, d);
  }
  red[tid] = lmax; __syncthreads();
  for (int s = 128; s > 0; s >>= 1) { if (tid < s) red[tid] = fmaxf(red[tid], red[tid + s]); __syncthreads(); }
  float gmax = red[0]; __syncthreads();

  float lsum = 0.0f, ltail = 0.0f;
  for (int idx = tid; idx < NVALID; idx += 256) {
    float e = __expf(sc[idx] - gmax);
    lsum += e;
    if (idx >= SB + 1) ltail += e;     // keys 1025..2047
  }
  red[tid] = lsum; __syncthreads();
  for (int s = 128; s > 0; s >>= 1) { if (tid < s) red[tid] += red[tid + s]; __syncthreads(); }
  float gsum = red[0]; __syncthreads();
  red[tid] = ltail; __syncthreads();
  for (int s = 128; s > 0; s >>= 1) { if (tid < s) red[tid] += red[tid + s]; __syncthreads(); }

  if (tid == 0) {
    float gtail = red[0];
    float pe   = __expf(sc[SB] - gmax) / gsum;             // probs at key e=1024
    float mean = (gtail / gsum) / 1023.0f + 1e-6f;
    float p = pe / mean;
    p = (p != p) ? 0.0f : fminf(fmaxf(p, 0.0f), 1.0f);
    float m = (bern[h] < p) ? 1.0f : 0.0f;
    sh_coef = m / (float)RB;
  }
  __syncthreads();
  if (tid < HDIM) ve[tid] = V[((size_t)h * T_SEQ + RB) * HDIM + tid] * sh_coef;
  __syncthreads();
  for (int idx = tid; idx < (T_SEQ - RB - 1) * HDIM; idx += 256) {
    int row = RB + 1 + (idx >> 7);
    int d   = idx & 127;
    V[((size_t)h * T_SEQ + row) * HDIM + d] += ve[d];
  }
}

// ---------------- kernel 5: flash attention, bf16x3, 32-key tiles ----------------
__global__ __launch_bounds__(128)
void attn_kernel(const float* __restrict__ Q, const __bf16* __restrict__ Khi,
                 const __bf16* __restrict__ Klo, const float* __restrict__ V,
                 float* __restrict__ Oout) {
  __shared__ __bf16 Qhi_s[64 * 136], Qlo_s[64 * 136];   // 64 queries x 128 dims
  __shared__ __bf16 Vhi_s[128 * 40], Vlo_s[128 * 40];   // [d][key], 32 keys
  __shared__ __bf16 Phi_s[4 * 16 * 32], Plo_s[4 * 16 * 32];

  const int tid  = threadIdx.x;
  const int w    = tid >> 5;         // 4 waves, 16 queries each
  const int lane = tid & 31;
  const int ln   = lane & 15;
  const int hf   = lane >> 4;
  const int h    = blockIdx.y;
  const int t0   = blockIdx.x * 64;
  const size_t hk = (size_t)h * T_SEQ;

  const float* Qh = Q + hk * HDIM;
  const float* Vh = V + hk * HDIM;

  // stage + split this block's 64 query rows
  #pragma unroll
  for (int i = 0; i < 16; ++i) {
    int idx = tid + i * 128;
    int row = idx >> 5, kc = (idx & 31) * 4;
    float4 v = *(const float4*)(Qh + (size_t)(t0 + row) * HDIM + kc);
    v4bf hh, ll; split4(v, hh, ll);
    *(v4bf*)(&Qhi_s[row * 136 + kc]) = hh;
    *(v4bf*)(&Qlo_s[row * 136 + kc]) = ll;
  }

  v8f Oacc[8];
  float mrow[8], lrow[8], alphas[8];
  #pragma unroll
  for (int ct = 0; ct < 8; ++ct)
    #pragma unroll
    for (int r = 0; r < 8; ++r) Oacc[ct][r] = 0.0f;
  #pragma unroll
  for (int r = 0; r < 8; ++r) { mrow[r] = -1e30f; lrow[r] = 0.0f; }

  __bf16* Pw_hi = &Phi_s[w * 16 * 32];
  __bf16* Pw_lo = &Plo_s[w * 16 * 32];

  for (int it = 0; it < 39; ++it) {     // 7 tiles for [0,204), 32 for [1024,2048)
    const int k0 = (it < 7) ? it * 32 : (T_SEQ - RB) + (it - 7) * 32;
    __syncthreads();
    // stage V tile 32x128, split + transpose -> [d][key]
    #pragma unroll
    for (int i = 0; i < 32; ++i) {
      int idx = tid + i * 128;
      int kr = idx >> 7, d = idx & 127;
      float v = Vh[(size_t)(k0 + kr) * HDIM + d];
      __bf16 hh = (__bf16)v;
      Vhi_s[d * 40 + kr] = hh;
      Vlo_s[d * 40 + kr] = (__bf16)(v - (float)hh);
    }
    __syncthreads();

    // S(16x32) = Q(16x128) * K^T(128x32), two 16-col tiles, bf16x3
    v8f S0, S1;
    #pragma unroll
    for (int r = 0; r < 8; ++r) { S0[r] = 0.0f; S1[r] = 0.0f; }
    const size_t kr0 = (hk + k0 + ln) * HDIM;
    const size_t kr1 = (hk + k0 + 16 + ln) * HDIM;
    #pragma unroll
    for (int kc = 0; kc < HDIM; kc += 32) {
      int qb = (w * 16 + ln) * 136 + kc + 8 * hf;
      v16bf qh = ld_frag(&Qhi_s[qb]), ql = ld_frag(&Qlo_s[qb]);
      v16bf kh0 = ld_frag(Khi + kr0 + kc + 8 * hf);
      v16bf kl0 = ld_frag(Klo + kr0 + kc + 8 * hf);
      v16bf kh1 = ld_frag(Khi + kr1 + kc + 8 * hf);
      v16bf kl1 = ld_frag(Klo + kr1 + kc + 8 * hf);
      // interleave the two independent accumulators
      S0 = wmma_bf16(qh, kl0, S0);
      S1 = wmma_bf16(qh, kl1, S1);
      S0 = wmma_bf16(ql, kh0, S0);
      S1 = wmma_bf16(ql, kh1, S1);
      S0 = wmma_bf16(qh, kh0, S0);
      S1 = wmma_bf16(qh, kh1, S1);
    }

    const bool m0c = (it == 6) && ((k0 + ln) >= SB);   // keys 192..207 partial
    const bool m1c = (it == 6);                        // keys 208..223 all masked
    #pragma unroll
    for (int r = 0; r < 8; ++r) {
      float s0 = S0[r] * SCALE_QK; if (m0c) s0 = -1e30f;
      float s1 = S1[r] * SCALE_QK; if (m1c) s1 = -1e30f;
      float tmax  = fmaxf(rmax16(s0), rmax16(s1));
      float mnew  = fmaxf(mrow[r], tmax);
      float alpha = __expf(mrow[r] - mnew);
      float p0 = __expf(s0 - mnew);
      float p1 = __expf(s1 - mnew);
      float rs = rsum16(p0 + p1);
      lrow[r]   = lrow[r] * alpha + rs;
      mrow[r]   = mnew;
      alphas[r] = alpha;
      int prow = (r + 8 * hf) * 32;
      __bf16 h0 = (__bf16)p0;
      Pw_hi[prow + ln] = h0;       Pw_lo[prow + ln] = (__bf16)(p0 - (float)h0);
      __bf16 h1 = (__bf16)p1;
      Pw_hi[prow + 16 + ln] = h1;  Pw_lo[prow + 16 + ln] = (__bf16)(p1 - (float)h1);
    }
    #pragma unroll
    for (int ct = 0; ct < 8; ++ct)
      #pragma unroll
      for (int r = 0; r < 8; ++r) Oacc[ct][r] *= alphas[r];

    asm volatile("s_wait_dscnt 0" ::: "memory");

    // O(16x128) += P(16x32) * V(32x128), bf16x3; process ct in pairs so the
    // 3-product chains alternate between two independent accumulators
    const int pb = ln * 32 + 8 * hf;
    v16bf ph = ld_frag(&Pw_hi[pb]), pl = ld_frag(&Pw_lo[pb]);
    #pragma unroll
    for (int ct = 0; ct < 8; ct += 2) {
      int vb0 = ((ct + 0) * 16 + ln) * 40 + 8 * hf;
      int vb1 = ((ct + 1) * 16 + ln) * 40 + 8 * hf;
      v16bf vh0 = ld_frag(&Vhi_s[vb0]), vl0 = ld_frag(&Vlo_s[vb0]);
      v16bf vh1 = ld_frag(&Vhi_s[vb1]), vl1 = ld_frag(&Vlo_s[vb1]);
      Oacc[ct + 0] = wmma_bf16(ph, vl0, Oacc[ct + 0]);
      Oacc[ct + 1] = wmma_bf16(ph, vl1, Oacc[ct + 1]);
      Oacc[ct + 0] = wmma_bf16(pl, vh0, Oacc[ct + 0]);
      Oacc[ct + 1] = wmma_bf16(pl, vh1, Oacc[ct + 1]);
      Oacc[ct + 0] = wmma_bf16(ph, vh0, Oacc[ct + 0]);
      Oacc[ct + 1] = wmma_bf16(ph, vh1, Oacc[ct + 1]);
    }
  }

  // write [T][D]: col = h*128 + d
  #pragma unroll
  for (int ct = 0; ct < 8; ++ct)
    #pragma unroll
    for (int r = 0; r < 8; ++r) {
      int t = t0 + w * 16 + r + 8 * hf;
      int c = h * HDIM + ct * 16 + ln;
      Oout[(size_t)t * DMODEL + c] = Oacc[ct][r] / lrow[r];
    }
}

// ---------------- launch ----------------
extern "C" void kernel_launch(void* const* d_in, const int* in_sizes, int n_in,
                              void* d_out, int out_size, void* d_ws, size_t ws_size,
                              hipStream_t stream) {
  (void)in_sizes; (void)n_in; (void)out_size; (void)ws_size;
  const float* hidden = (const float*)d_in[0];
  const float* wq     = (const float*)d_in[1];
  const float* wk     = (const float*)d_in[2];
  const float* wv     = (const float*)d_in[3];
  const float* wo     = (const float*)d_in[4];
  const float* bern   = (const float*)d_in[5];

  const size_t NE = (size_t)T_SEQ * DMODEL;   // 4,194,304
  const int    NEi = (int)NE;
  float* ws    = (float*)d_ws;
  float* q     = ws;            // [H][T][128] f32
  float* k     = ws + 1 * NE;
  float* v     = ws + 2 * NE;
  float* attnb = ws + 3 * NE;   // [T][D]
  __bf16* xhi  = (__bf16*)(ws + 4 * NE);  // activation split (hidden, then attnb)
  __bf16* xlo  = xhi + NE;
  __bf16* whi  = xlo + NE;                // current weight split (reused 4x)
  __bf16* wlo  = whi + NE;
  __bf16* khi  = wlo + NE;                // RoPE'd K split
  __bf16* klo  = khi + NE;

  const int SGRID = (NEi + 255) / 256;
  dim3 ggrid(DMODEL / 128, T_SEQ / 64);       // (16, 32)

  // sanitized hidden -> bf16 hi/lo
  split_kernel<<<SGRID, 256, 0, stream>>>(hidden, xhi, xlo, NEi, 1);

  split_kernel<<<SGRID, 256, 0, stream>>>(wq, whi, wlo, NEi, 0);
  gemm_bf16x3_async_kernel<<<ggrid, 256, 0, stream>>>(xhi, xlo, whi, wlo, q, 0);
  split_kernel<<<SGRID, 256, 0, stream>>>(wk, whi, wlo, NEi, 0);
  gemm_bf16x3_async_kernel<<<ggrid, 256, 0, stream>>>(xhi, xlo, whi, wlo, k, 0);
  split_kernel<<<SGRID, 256, 0, stream>>>(wv, whi, wlo, NEi, 0);
  gemm_bf16x3_async_kernel<<<ggrid, 256, 0, stream>>>(xhi, xlo, whi, wlo, v, 0);

  int nrope = NHEAD * T_SEQ * 64;
  rope_split_kernel<<<(nrope + 255) / 256, 256, 0, stream>>>(q, k, khi, klo);

  vupdate_kernel<<<NHEAD, 256, 0, stream>>>(q, k, bern, v);

  attn_kernel<<<dim3(T_SEQ / 64, NHEAD), 128, 0, stream>>>(q, khi, klo, v, attnb);

  // output projection: split attnb (reuse x buffers) and wo, then GEMM -> d_out
  split_kernel<<<SGRID, 256, 0, stream>>>(attnb, xhi, xlo, NEi, 0);
  split_kernel<<<SGRID, 256, 0, stream>>>(wo, whi, wlo, NEi, 0);
  gemm_bf16x3_async_kernel<<<ggrid, 256, 0, stream>>>(xhi, xlo, whi, wlo, (float*)d_out, 1);
}
